// DeepAR_24550033064386
// MI455X (gfx1250) — matrix operand order repaired and checked by reading, not compiled, not run on hardware
//
#include <hip/hip_runtime.h>

typedef __attribute__((ext_vector_type(16))) __bf16 v16bf;
typedef __attribute__((ext_vector_type(8)))  float  v8f;

union Frag { v16bf v; uint4 q[2]; };

#define ROWS_WG   64
#define N_GATES   160
#define KDIM      64
#define T_COND    167
#define T_TOTAL   191

__device__ __forceinline__ unsigned short f2bf(float f) {
  unsigned u = __float_as_uint(f);
  u += 0x7fffu + ((u >> 16) & 1u);          // round-to-nearest-even
  return (unsigned short)(u >> 16);
}
__device__ __forceinline__ float bf2f(unsigned short h) {
  return __uint_as_float(((unsigned)h) << 16);
}
// branchless sigmoid: one v_exp_f32 + one v_rcp_f32
__device__ __forceinline__ float sigm(float x) {
  return __builtin_amdgcn_rcpf(1.0f + __expf(-x));
}
// branchless tanh: t=exp(-2|x|); (1-t)/(1+t) with sign re-applied.
// Stable for all x (|x| large -> t -> 0 -> r -> 1). No exec-mask divergence.
__device__ __forceinline__ float fast_tanh(float x) {
  float ax = __builtin_fabsf(x);
  float t  = __expf(-2.0f * ax);
  float r  = (1.0f - t) * __builtin_amdgcn_rcpf(1.0f + t);
  return __uint_as_float(__float_as_uint(r) | (__float_as_uint(x) & 0x80000000u));
}

__device__ __forceinline__ float gaussz(unsigned x) {
  x ^= x >> 17; x *= 0xed5ad4bbu; x ^= x >> 11;
  x *= 0xac4c1b51u; x ^= x >> 15; x *= 0x31848babu; x ^= x >> 14;
  unsigned y = x * 0x9E3779B9u; y ^= y >> 16;
  float u1 = ((x >> 9) + 1u) * (1.0f / 8388609.0f);
  float u2 = (y >> 9) * (1.0f / 8388608.0f);
  return __fsqrt_rn(-2.0f * __logf(u1)) * __cosf(6.28318530718f * u2);
}

__global__ __launch_bounds__(256) void deepar_lstm_kernel(
    const float* __restrict__ z1,    const float* __restrict__ xc,
    const float* __restrict__ W_emb, const float* __restrict__ b_emb,
    const float* __restrict__ W_ih,  const float* __restrict__ b_ih,
    const float* __restrict__ W_hh,  const float* __restrict__ b_hh,
    const float* __restrict__ W_mu,  const float* __restrict__ b_mu,
    const float* __restrict__ W_sig, const float* __restrict__ b_sig,
    float* __restrict__ out)
{
  // gates buffer (f32) aliased with the bf16 weight staging area: weights
  // (20.5 KB) are only read into registers before the time loop.
  __shared__ float sGates[ROWS_WG][N_GATES];       // 40960 B
  __shared__ unsigned short sHX[ROWS_WG][KDIM];    //  8192 B  [h(40)|emb(20)|xc(3)|0]
  __shared__ float sBtot[N_GATES];
  __shared__ float sWemb[20], sBemb[20], sWmu[40], sWsig[40];
  __shared__ float sV[ROWS_WG], sVinv[ROWS_WG], sY[ROWS_WG];
  __shared__ float sScal[2];

  unsigned short (*sW)[KDIM] = (unsigned short(*)[KDIM]) &sGates[0][0];

  const int tid   = threadIdx.x;
  const int lane  = tid & 31;
  const int wave  = tid >> 5;
  const int lmod  = lane & 15;
  const int lhalf = lane >> 4;
  const int rowBase = (wave & 3) * 16;   // 4 row-groups of 16
  const int nBase   = (wave >> 2) * 80;  // 2 N-halves of 80 gates
  const long wgRow0 = (long)blockIdx.x * ROWS_WG;

  // ---------------- init: stage weights to LDS (bf16), state, stats ----------------
  for (int i = tid; i < N_GATES * KDIM; i += 256) {
    int g = i >> 6, k = i & 63;
    float w = (k < 40) ? W_hh[g * 40 + k]
            : ((k < 63) ? W_ih[g * 23 + (k - 40)] : 0.0f);
    sW[g][k] = f2bf(w);
  }
  for (int i = tid; i < ROWS_WG * KDIM; i += 256) sHX[i >> 6][i & 63] = 0;
  for (int i = tid; i < N_GATES; i += 256) sBtot[i] = b_ih[i] + b_hh[i];
  if (tid < 20) { sWemb[tid] = W_emb[tid]; sBemb[tid] = b_emb[tid]; }
  if (tid >= 32 && tid < 72) { sWmu[tid - 32] = W_mu[tid - 32]; sWsig[tid - 32] = W_sig[tid - 32]; }
  if (tid == 0) { sScal[0] = b_mu[0]; sScal[1] = b_sig[0]; }
  if (tid < ROWS_WG) {
    long b = wgRow0 + tid;
    const float* zp = z1 + b * 168;
    float s = 0.f;
    for (int k = 0; k < 168; ++k) s += zp[k];
    float v = s * (1.0f / 168.0f) + 1.0f;   // v == v2
    sV[tid]    = v;
    sVinv[tid] = 1.0f / v;                  // full-precision divide, once
    sY[tid]    = zp[167];                   // y0 = z1[:, -1]
  }
  __syncthreads();

  // ---------------- preload all B (weight) fragments into registers ----------------
  // bf16 B 32x16 layout: lanes 0-15 K{0-7,16-23}, lanes 16-31 K{8-15,24-31}
  Frag Bf[5][2];
#pragma unroll
  for (int nt = 0; nt < 5; ++nt) {
    int n = nBase + nt * 16 + lmod;
#pragma unroll
    for (int kc = 0; kc < 2; ++kc) {
      const unsigned short* p = &sW[n][kc * 32 + lhalf * 8];
      Bf[nt][kc].q[0] = *(const uint4*)(p);
      Bf[nt][kc].q[1] = *(const uint4*)(p + 16);
    }
  }
  __syncthreads();   // weight staging area may now be reused as sGates

  float creg[10];
#pragma unroll
  for (int q = 0; q < 10; ++q) creg[q] = 0.f;
  const int crow  = tid >> 2;         // cell row, 0..63
  const int jbase = (tid & 3) * 10;   // 10 hidden units per thread

  // =============================== time loop ===============================
  for (int t = 0; t < T_TOTAL; ++t) {
    // --- phase A: per-row scalar embedding + covariates into sHX cols 40..62 ---
    if (tid < ROWS_WG) {
      long b = wgRow0 + tid;
      float s = ((t < T_COND) ? z1[b * 168 + t] : sY[tid]) * sVinv[tid];
#pragma unroll
      for (int e = 0; e < 20; ++e)
        sHX[tid][40 + e] = f2bf(s * sWemb[e] + sBemb[e]);
      const float* xr = xc + (b * 192 + (t + 1)) * 3;
      sHX[tid][60] = f2bf(xr[0]);
      sHX[tid][61] = f2bf(xr[1]);
      sHX[tid][62] = f2bf(xr[2]);
    }
    __syncthreads();

    // --- GEMM: gates = [h|emb|xc] @ [W_hh|W_ih]^T via v_wmma_f32_16x16x32_bf16 ---
    // bf16 A 16x32 layout: lanes 0-15 K{0-7,16-23}, lanes 16-31 K{8-15,24-31}
    Frag A0, A1;
    {
      const unsigned short* ar = &sHX[rowBase + lmod][0];
      A0.q[0] = *(const uint4*)(ar + lhalf * 8);
      A0.q[1] = *(const uint4*)(ar + 16 + lhalf * 8);
      A1.q[0] = *(const uint4*)(ar + 32 + lhalf * 8);
      A1.q[1] = *(const uint4*)(ar + 48 + lhalf * 8);
    }
    v8f acc[5];
#pragma unroll
    for (int nt = 0; nt < 5; ++nt) {
      v8f c = {};
      c = __builtin_amdgcn_wmma_f32_16x16x32_bf16(false, A0.v, false, Bf[nt][0].v,
                                                  (short)0, c, false, false);
      c = __builtin_amdgcn_wmma_f32_16x16x32_bf16(false, A1.v, false, Bf[nt][1].v,
                                                  (short)0, c, false, false);
      acc[nt] = c;
    }
    // scatter D (32-bit C/D layout: VGPR r -> M=r (+8 for lanes>=16), N=lane&15)
#pragma unroll
    for (int nt = 0; nt < 5; ++nt) {
      int n  = nBase + nt * 16 + lmod;
      int m0 = rowBase + lhalf * 8;
#pragma unroll
      for (int r = 0; r < 8; ++r)
        sGates[m0 + r][n] = acc[nt][r];
    }
    __syncthreads();

    // --- LSTM cell: c = sig(f)*c + sig(i)*tanh(g); h = sig(o)*tanh(c) ---
    // fully branchless: v_exp_f32 / v_rcp_f32 only, no exec-mask divergence
#pragma unroll
    for (int q = 0; q < 10; ++q) {
      int j = jbase + q;
      float ig = sGates[crow][j]        + sBtot[j];
      float fg = sGates[crow][40 + j]   + sBtot[40 + j];
      float gg = sGates[crow][80 + j]   + sBtot[80 + j];
      float og = sGates[crow][120 + j]  + sBtot[120 + j];
      float c  = sigm(fg) * creg[q] + sigm(ig) * fast_tanh(gg);
      creg[q]  = c;
      sHX[crow][j] = f2bf(sigm(og) * fast_tanh(c));
    }

    // --- prediction head + sampling (t is block-uniform; barrier is safe) ---
    if (t >= T_COND) {
      __syncthreads();
      if (tid < ROWS_WG) {
        int p  = t - T_COND;
        long b = wgRow0 + tid;
        float mu = sScal[0], sg = sScal[1];
#pragma unroll
        for (int j = 0; j < 40; ++j) {
          float h = bf2f(sHX[tid][j]);
          mu += h * sWmu[j];
          sg += h * sWsig[j];
        }
        sg = (sg > 20.f) ? sg : log1pf(__expf(sg));   // softplus
        float eps = gaussz((unsigned)(b * 24 + p) * 0x9E3779B9u + 0x85ebca6bu);
        float y = mu + sg * eps;
        sY[tid] = y;
        float v2 = sV[tid];
        out[b * 24 + p]          = y  * v2;   // outputs
        out[196608 + b * 24 + p] = mu * v2;   // mus
        out[393216 + b * 24 + p] = sg * v2;   // sigmas
      }
    }
    __syncthreads();
  }
}

extern "C" void kernel_launch(void* const* d_in, const int* in_sizes, int n_in,
                              void* d_out, int out_size, void* d_ws, size_t ws_size,
                              hipStream_t stream) {
  (void)in_sizes; (void)n_in; (void)out_size; (void)d_ws; (void)ws_size;
  deepar_lstm_kernel<<<128, 256, 0, stream>>>(
      (const float*)d_in[0],  (const float*)d_in[1],
      (const float*)d_in[2],  (const float*)d_in[3],
      (const float*)d_in[4],  (const float*)d_in[5],
      (const float*)d_in[6],  (const float*)d_in[7],
      (const float*)d_in[8],  (const float*)d_in[9],
      (const float*)d_in[10], (const float*)d_in[11],
      (float*)d_out);
}
